// MultiHeadedAttention_8950711845863
// MI455X (gfx1250) — matrix operand order repaired and checked by reading, not compiled, not run on hardware
//
#include <hip/hip_runtime.h>

#define B_  8
#define T_  1024
#define F_  1024
#define H_  16
#define DK_ 64

typedef __attribute__((ext_vector_type(16))) _Float16 v16h;
typedef __attribute__((ext_vector_type(4)))  _Float16 v4h;
typedef __attribute__((ext_vector_type(8)))  float    v8f;
typedef __attribute__((ext_vector_type(4)))  float    v4f;
typedef __attribute__((ext_vector_type(4)))  int      v4i;

#define AS1 __attribute__((address_space(1)))
#define AS3 __attribute__((address_space(3)))

#if defined(__has_builtin)
#if __has_builtin(__builtin_amdgcn_global_load_async_to_lds_b128)
#define USE_ASYNC_LDS 1
#endif
#endif
#ifndef USE_ASYNC_LDS
#define USE_ASYNC_LDS 0
#endif

union FragU { v16h h; v4i i[2]; };

__device__ __forceinline__ v8f wmma32(v16h a, v16h b, v8f c) {
    return __builtin_amdgcn_wmma_f32_16x16x32_f16(false, a, false, b, (short)0, c,
                                                  false, false);
}

// A fragment (16x32 f16, M x K), source row-major [M][K]  (ISA 7.12.2)
__device__ __forceinline__ v16h frag_a(const _Float16* p, int ld, int m0, int k0,
                                       int lo, int hi) {
    const _Float16* r = p + (size_t)(m0 + lo) * ld + k0 + hi * 8;
    FragU u;
    u.i[0] = *(const v4i*)(r);
    u.i[1] = *(const v4i*)(r + 16);
    return u.h;
}

// B fragment (32x16 f16, K x N), source row-major [N][K]
__device__ __forceinline__ v16h frag_b(const _Float16* p, int ld, int n0, int k0,
                                       int lo, int hi) {
    const _Float16* r = p + (size_t)(n0 + lo) * ld + k0 + hi * 8;
    FragU u;
    u.i[0] = *(const v4i*)(r);
    u.i[1] = *(const v4i*)(r + 16);
    return u.h;
}

template <int N>
__device__ __forceinline__ void wait_async() {
#if defined(__has_builtin) && __has_builtin(__builtin_amdgcn_s_wait_asynccnt)
    __builtin_amdgcn_s_wait_asynccnt(N);
#else
    if constexpr (N == 0) asm volatile("s_wait_asynccnt 0x0" ::: "memory");
    else                  asm volatile("s_wait_asynccnt 0x2" ::: "memory");
#endif
}

// ---------------------------------------------------------------------------
// QKV projection: out = X @ W^T + b, X (8192,1024) f32, W (1024,1024) f32.
// Block = 256 thr (8 waves), tile 128x128, wave tile 32x64, K-step 32.
// MODE 0: q -> (B,H,T,DK) f16 * 1/sqrt(DK); 1: k -> (B,H,T,DK); 2: v -> (B,H,DK,T)
// ---------------------------------------------------------------------------
template <int MODE>
__global__ __launch_bounds__(256) void proj_qkv(const float* __restrict__ X,
                                                const float* __restrict__ W,
                                                const float* __restrict__ bias,
                                                _Float16* __restrict__ out) {
    __shared__ __align__(16) _Float16 sA[128 * 32];
    __shared__ __align__(16) _Float16 sB[128 * 32];
    const int tid  = threadIdx.x;
    const int lane = tid & 31, wv = tid >> 5;
    const int lo = lane & 15, hi = lane >> 4;
    const int wm = wv >> 1, wn = wv & 1;
    const int m_blk = blockIdx.x * 128;
    const int n_blk = blockIdx.y * 128;

    v8f acc[2][4] = {};

    for (int kk = 0; kk < F_; kk += 32) {
#pragma unroll
        for (int it = 0; it < 4; ++it) {
            int idx = tid + it * 256;
            int row = idx >> 3, c = (idx & 7) * 4;
            v4f fa = *(const v4f*)(X + (size_t)(m_blk + row) * F_ + kk + c);
            v4h ha = {(_Float16)fa[0], (_Float16)fa[1], (_Float16)fa[2], (_Float16)fa[3]};
            *(v4h*)(sA + row * 32 + c) = ha;
            v4f fb = *(const v4f*)(W + (size_t)(n_blk + row) * F_ + kk + c);
            v4h hb = {(_Float16)fb[0], (_Float16)fb[1], (_Float16)fb[2], (_Float16)fb[3]};
            *(v4h*)(sB + row * 32 + c) = hb;
        }
        __syncthreads();
        v16h af0 = frag_a(sA, 32, wm * 32, 0, lo, hi);
        v16h af1 = frag_a(sA, 32, wm * 32 + 16, 0, lo, hi);
#pragma unroll
        for (int j = 0; j < 4; ++j) {
            v16h bf = frag_b(sB, 32, wn * 64 + j * 16, 0, lo, hi);
            acc[0][j] = wmma32(af0, bf, acc[0][j]);
            acc[1][j] = wmma32(af1, bf, acc[1][j]);
        }
        __syncthreads();
    }

#pragma unroll
    for (int j = 0; j < 4; ++j) {
        int ng = n_blk + wn * 64 + j * 16 + lo;
        float bj = bias[ng];
        int hh = ng >> 6, dk = ng & 63;
#pragma unroll
        for (int i = 0; i < 2; ++i) {
#pragma unroll
            for (int r = 0; r < 8; ++r) {
                int mg = m_blk + wm * 32 + i * 16 + r + hi * 8;
                int bb = mg >> 10, tt = mg & 1023;
                float v = acc[i][j][r] + bj;
                if (MODE == 0) v *= 0.125f;  // 1/sqrt(64)
                size_t o;
                if (MODE == 2)
                    o = (((size_t)bb * H_ + hh) * DK_ + dk) * T_ + tt;
                else
                    o = (((size_t)bb * H_ + hh) * T_ + tt) * DK_ + dk;
                out[o] = (_Float16)v;
            }
        }
    }
}

// ---------------------------------------------------------------------------
// Output projection: out = (ctx @ Wo^T + bo) * mask, ctx f16, out f32
// ---------------------------------------------------------------------------
__global__ __launch_bounds__(256) void proj_out(const _Float16* __restrict__ Xh,
                                                const float* __restrict__ W,
                                                const float* __restrict__ bias,
                                                const float* __restrict__ mask,
                                                float* __restrict__ out) {
    __shared__ __align__(16) _Float16 sA[128 * 32];
    __shared__ __align__(16) _Float16 sB[128 * 32];
    const int tid  = threadIdx.x;
    const int lane = tid & 31, wv = tid >> 5;
    const int lo = lane & 15, hi = lane >> 4;
    const int wm = wv >> 1, wn = wv & 1;
    const int m_blk = blockIdx.x * 128;
    const int n_blk = blockIdx.y * 128;

    v8f acc[2][4] = {};

    for (int kk = 0; kk < F_; kk += 32) {
#pragma unroll
        for (int it = 0; it < 4; ++it) {
            int idx = tid + it * 256;
            int row = idx >> 3, c = (idx & 7) * 4;
            *(v4h*)(sA + row * 32 + c) =
                *(const v4h*)(Xh + (size_t)(m_blk + row) * F_ + kk + c);
            v4f fb = *(const v4f*)(W + (size_t)(n_blk + row) * F_ + kk + c);
            v4h hb = {(_Float16)fb[0], (_Float16)fb[1], (_Float16)fb[2], (_Float16)fb[3]};
            *(v4h*)(sB + row * 32 + c) = hb;
        }
        __syncthreads();
        v16h af0 = frag_a(sA, 32, wm * 32, 0, lo, hi);
        v16h af1 = frag_a(sA, 32, wm * 32 + 16, 0, lo, hi);
#pragma unroll
        for (int j = 0; j < 4; ++j) {
            v16h bf = frag_b(sB, 32, wn * 64 + j * 16, 0, lo, hi);
            acc[0][j] = wmma32(af0, bf, acc[0][j]);
            acc[1][j] = wmma32(af1, bf, acc[1][j]);
        }
        __syncthreads();
    }

#pragma unroll
    for (int j = 0; j < 4; ++j) {
        int ng = n_blk + wn * 64 + j * 16 + lo;
        float bj = bias[ng];
#pragma unroll
        for (int i = 0; i < 2; ++i) {
#pragma unroll
            for (int r = 0; r < 8; ++r) {
                int mg = m_blk + wm * 32 + i * 16 + r + hi * 8;
                out[(size_t)mg * F_ + ng] = (acc[i][j][r] + bj) * mask[mg];
            }
        }
    }
}

// ---------------------------------------------------------------------------
// Flash attention: grid (T/128, B*H), 8 waves/block, wave = 16 queries.
// K/V tiles staged cooperatively into LDS via async-to-LDS (double buffered),
// tracked on ASYNCcnt; falls back to direct global fragment loads.
// ---------------------------------------------------------------------------
#if USE_ASYNC_LDS
__device__ __forceinline__ void stage_kv_async(const _Float16* kp, const _Float16* vp,
                                               int kt, _Float16* dK, _Float16* dV,
                                               int tid) {
    // K tile: 32 keys x 64 dk = 4KB, 256 lanes x 16B
    int kr = tid >> 3, kc = (tid & 7) * 8;
    __builtin_amdgcn_global_load_async_to_lds_b128(
        (AS1 v4i*)(AS1 void*)(kp + (size_t)(kt + kr) * DK_ + kc),
        (AS3 v4i*)(AS3 void*)(dK + kr * DK_ + kc), 0, 0);
    // V^T tile: 64 dk x 32 keys = 4KB
    int vr = tid >> 2, vc = (tid & 3) * 8;
    __builtin_amdgcn_global_load_async_to_lds_b128(
        (AS1 v4i*)(AS1 void*)(vp + (size_t)vr * T_ + kt + vc),
        (AS3 v4i*)(AS3 void*)(dV + vr * 32 + vc), 0, 0);
}
#endif

__global__ __launch_bounds__(256) void attn(const _Float16* __restrict__ Q,
                                            const _Float16* __restrict__ K,
                                            const _Float16* __restrict__ VT,
                                            _Float16* __restrict__ ctx) {
    __shared__ __align__(16) _Float16 sP[8][16 * 32];
#if USE_ASYNC_LDS
    __shared__ __align__(16) _Float16 sK[2][32 * 64];
    __shared__ __align__(16) _Float16 sV[2][64 * 32];
#endif
    const int tid  = threadIdx.x;
    const int lane = tid & 31, wv = tid >> 5;
    const int lo = lane & 15, hi = lane >> 4;
    const int bh = blockIdx.y;
    const int bb = bh >> 4, hh = bh & 15;
    const int q0 = blockIdx.x * 128 + wv * 16;
    const _Float16* qp = Q + (size_t)bh * T_ * DK_;
    const _Float16* kp = K + (size_t)bh * T_ * DK_;
    const _Float16* vp = VT + (size_t)bh * DK_ * T_;
    _Float16* myP = &sP[wv][0];

    v16h aq0 = frag_a(qp, DK_, q0, 0, lo, hi);
    v16h aq1 = frag_a(qp, DK_, q0, 32, lo, hi);

    float mrow[8], lrow[8];
    v8f oacc[4] = {};
#pragma unroll
    for (int r = 0; r < 8; ++r) { mrow[r] = -3.0e38f; lrow[r] = 0.0f; }

#if USE_ASYNC_LDS
    int buf = 0;
    stage_kv_async(kp, vp, 0, sK[0], sV[0], tid);
#endif

    for (int kt = 0; kt < T_; kt += 32) {
#if USE_ASYNC_LDS
        const bool pf = (kt + 32) < T_;
        if (pf) stage_kv_async(kp, vp, kt + 32, sK[buf ^ 1], sV[buf ^ 1], tid);
        if (pf) wait_async<2>(); else wait_async<0>();
        __syncthreads();
        const _Float16* cK = sK[buf];
        const _Float16* cV = sV[buf];
        const int kbase = 0, vk = 0, ldv = 32;
#else
        const _Float16* cK = kp;
        const _Float16* cV = vp;
        const int kbase = kt, vk = kt, ldv = T_;
#endif
        // S = Q (16x64) @ K^T (64x32): two N-tiles, two k-steps each
        v8f s0 = {}, s1 = {};
        {
            v16h b00 = frag_b(cK, DK_, kbase, 0, lo, hi);
            v16h b10 = frag_b(cK, DK_, kbase + 16, 0, lo, hi);
            s0 = wmma32(aq0, b00, s0);
            s1 = wmma32(aq0, b10, s1);
            v16h b01 = frag_b(cK, DK_, kbase, 32, lo, hi);
            v16h b11 = frag_b(cK, DK_, kbase + 16, 32, lo, hi);
            s0 = wmma32(aq1, b01, s0);
            s1 = wmma32(aq1, b11, s1);
        }
        // online softmax: row (= r + hi*8) lives in one 16-lane half
        float rmax[8];
#pragma unroll
        for (int r = 0; r < 8; ++r) rmax[r] = fmaxf(s0[r], s1[r]);
#pragma unroll
        for (int off = 1; off < 16; off <<= 1)
#pragma unroll
            for (int r = 0; r < 8; ++r)
                rmax[r] = fmaxf(rmax[r], __shfl_xor(rmax[r], off, 32));

        float e0[8], e1[8], psum[8];
#pragma unroll
        for (int r = 0; r < 8; ++r) {
            float mnew = fmaxf(mrow[r], rmax[r]);
            float sc = __expf(mrow[r] - mnew);
            e0[r] = __expf(s0[r] - mnew);
            e1[r] = __expf(s1[r] - mnew);
            mrow[r] = mnew;
            psum[r] = e0[r] + e1[r];
            lrow[r] *= sc;
#pragma unroll
            for (int n = 0; n < 4; ++n) oacc[n][r] *= sc;
        }
#pragma unroll
        for (int off = 1; off < 16; off <<= 1)
#pragma unroll
            for (int r = 0; r < 8; ++r) psum[r] += __shfl_xor(psum[r], off, 32);
#pragma unroll
        for (int r = 0; r < 8; ++r) lrow[r] += psum[r];

        // stage P (16x32 f16) to LDS in [M][K] order, reload as A fragment
#pragma unroll
        for (int r = 0; r < 8; ++r) {
            myP[(r + hi * 8) * 32 + lo]      = (_Float16)e0[r];
            myP[(r + hi * 8) * 32 + 16 + lo] = (_Float16)e1[r];
        }
        asm volatile("s_wait_dscnt 0" ::: "memory");
        v16h ap = frag_a(myP, 32, 0, 0, lo, hi);
#pragma unroll
        for (int n = 0; n < 4; ++n) {
            v16h bv = frag_b(cV, ldv, n * 16, vk, lo, hi);
            oacc[n] = wmma32(ap, bv, oacc[n]);
        }
#if USE_ASYNC_LDS
        __syncthreads();  // all waves done reading buf before it is refilled
        buf ^= 1;
#endif
    }

#pragma unroll
    for (int n = 0; n < 4; ++n)
#pragma unroll
        for (int r = 0; r < 8; ++r) {
            float v = oacc[n][r] / lrow[r];
            size_t o = ((size_t)(bb * T_ + q0 + r + hi * 8)) * F_ + hh * DK_ + n * 16 + lo;
            ctx[o] = (_Float16)v;
        }
}

extern "C" void kernel_launch(void* const* d_in, const int* in_sizes, int n_in,
                              void* d_out, int out_size, void* d_ws, size_t ws_size,
                              hipStream_t stream) {
    const float* query = (const float*)d_in[0];
    const float* key_  = (const float*)d_in[1];
    const float* value = (const float*)d_in[2];
    const float* mask  = (const float*)d_in[3];
    const float* Wq = (const float*)d_in[4];
    const float* bq = (const float*)d_in[5];
    const float* Wk = (const float*)d_in[6];
    const float* bk = (const float*)d_in[7];
    const float* Wv = (const float*)d_in[8];
    const float* bv = (const float*)d_in[9];
    const float* Wo = (const float*)d_in[10];
    const float* bo = (const float*)d_in[11];
    float* out = (float*)d_out;

    const size_t n_elem = (size_t)B_ * H_ * T_ * DK_;  // 8M f16 per tensor
    _Float16* qws  = (_Float16*)d_ws;
    _Float16* kws  = qws + n_elem;
    _Float16* vtws = kws + n_elem;
    _Float16* ctx  = vtws + n_elem;

    dim3 blk(256);
    dim3 gp(64, 8);           // 8192/128 x 1024/128
    proj_qkv<0><<<gp, blk, 0, stream>>>(query, Wq, bq, qws);
    proj_qkv<1><<<gp, blk, 0, stream>>>(key_, Wk, bk, kws);
    proj_qkv<2><<<gp, blk, 0, stream>>>(value, Wv, bv, vtws);
    dim3 ga(8, 128);          // T/128 x B*H
    attn<<<ga, blk, 0, stream>>>(qws, kws, vtws, ctx);
    proj_out<<<gp, blk, 0, stream>>>(ctx, Wo, bo, mask, out);
}
